// SubgraphSelector_58428735095403
// MI455X (gfx1250) — matrix operand order repaired and checked by reading, not compiled, not run on hardware
//
#include <hip/hip_runtime.h>
#include <hip/hip_bf16.h>
#include <math.h>

#define HIDDEN 128

typedef __attribute__((ext_vector_type(16))) __bf16 v16bf;
typedef __attribute__((ext_vector_type(8)))  __bf16 v8bf;
typedef __attribute__((ext_vector_type(8)))  float  v8f;

union V16U { v16bf v; v8bf h[2]; };

// ---------- prep kernel A: h (f32) -> h_bf (bf16), 6.4M elements ----------
__global__ void convert_h_bf16(const float* __restrict__ h,
                               __bf16* __restrict__ h_bf, int n) {
    int i = blockIdx.x * blockDim.x + threadIdx.x;
    if (i < n) h_bf[i] = (__bf16)h[i];
}

// ---------- prep kernel B: pack W1 [256,128] f32 into WMMA B-fragment layout ----------
// Packed element t = ((kt*8+nt)*32 + lane)*16 + i  holds
//   W1[K][N] with K = kt*32 + 8*half + (i&7) + 16*(i>>3), N = nt*16 + (lane&15), half = lane>>4
// (matches A table order: lanes 0-15: K=0..7,16..23, lanes 16-31: K=8..15,24..31,
//  two K per VGPR). Each lane then loads its 16 bf16 as two contiguous b128.
__global__ void pack_w1(const float* __restrict__ W1, __bf16* __restrict__ w1p) {
    int t = blockIdx.x * blockDim.x + threadIdx.x;   // 0 .. 64*32*16-1
    int i    = t & 15;
    int lane = (t >> 4) & 31;
    int tile = t >> 9;                                // 0..63 = kt*8+nt
    int nt   = tile & 7;
    int kt   = tile >> 3;
    int half = lane >> 4;
    int K = kt * 32 + 8 * half + (i & 7) + 16 * (i >> 3);
    int N = nt * 16 + (lane & 15);
    w1p[t] = (__bf16)W1[K * HIDDEN + N];
}

// ---------- main kernel: fused gather + GEMM(WMMA) + bias/ReLU + W2 dot + sigmoid ----
__global__ __launch_bounds__(256)
void edge_mlp_wmma(const __bf16* __restrict__ h_bf,
                   const int*    __restrict__ eidx,   // [2*E]
                   const __bf16* __restrict__ w1p,
                   const float*  __restrict__ b1,
                   const float*  __restrict__ W2,
                   const float*  __restrict__ b2,
                   float*        __restrict__ out,
                   int E) {
    const int lane = threadIdx.x & 31;
    const int wave = threadIdx.x >> 5;
    const int m    = lane & 15;      // row within M-tile / column within N-tile
    const int half = lane >> 4;

    const long long base = ((long long)blockIdx.x * 8 + wave) * 32;

    // Per-lane epilogue params: this lane owns column n = nt*16 + m of each N-tile.
    float b1v[8], w2v[8];
#pragma unroll
    for (int nt = 0; nt < 8; ++nt) {
        b1v[nt] = b1[nt * 16 + m];
        w2v[nt] = W2[nt * 16 + m];
    }

    // Gather feature-row pointers for the two 16-edge M-tiles (clamped, no branches
    // before the WMMAs so EXEC stays all-ones).
    long long e0 = base + m;
    long long e1 = base + 16 + m;
    long long e0c = (e0 < E) ? e0 : 0;
    long long e1c = (e1 < E) ? e1 : 0;
    const __bf16* rowS0 = h_bf + (long long)eidx[e0c] * HIDDEN;
    const __bf16* rowD0 = h_bf + (long long)eidx[(long long)E + e0c] * HIDDEN;
    const __bf16* rowS1 = h_bf + (long long)eidx[e1c] * HIDDEN;
    const __bf16* rowD1 = h_bf + (long long)eidx[(long long)E + e1c] * HIDDEN;

    v8f acc0[8], acc1[8];
#pragma unroll
    for (int nt = 0; nt < 8; ++nt) { v8f z = {}; acc0[nt] = z; acc1[nt] = z; }

    // K = 256 in 8 steps of 32 (kt 0..3 -> src half, kt 4..7 -> dst half).
    // A-fragment (ISA 16-bit A 16x32): lane(half h) elements 0-7 = K 8h+0..7,
    // elements 8-15 = K 16+8h+0..7  -> two contiguous b128 loads from the row.
    // Explicit one-ahead double buffering of A fragments (per kt) and B tiles
    // (per flattened (kt,nt) index) so WMMAs wait on loadcnt<=inflight, not 0.
    V16U A0[2], A1[2], B[2];

#define LOAD_A(ktv, x0, x1)                                            \
    do {                                                               \
        const __bf16* r0_ = ((ktv) < 4) ? rowS0 : rowD0;               \
        const __bf16* r1_ = ((ktv) < 4) ? rowS1 : rowD1;               \
        const int ko_ = ((ktv) & 3) * 32 + 8 * half;                   \
        (x0).h[0] = *(const v8bf*)(r0_ + ko_);                         \
        (x0).h[1] = *(const v8bf*)(r0_ + ko_ + 16);                    \
        (x1).h[0] = *(const v8bf*)(r1_ + ko_);                         \
        (x1).h[1] = *(const v8bf*)(r1_ + ko_ + 16);                    \
    } while (0)

#define LOAD_B(idxv, bx)                                               \
    do {                                                               \
        const __bf16* bp_ = w1p + (((idxv) * 32 + lane) << 4);         \
        (bx).h[0] = *(const v8bf*)(bp_);                               \
        (bx).h[1] = *(const v8bf*)(bp_ + 8);                           \
    } while (0)

    LOAD_A(0, A0[0], A1[0]);
    LOAD_B(0, B[0]);

#pragma unroll
    for (int idx = 0; idx < 64; ++idx) {
        const int kt = idx >> 3;
        const int nt = idx & 7;
        const int ka = kt & 1;
        const int bb = idx & 1;
        if (nt == 0 && kt + 1 < 8) LOAD_A(kt + 1, A0[ka ^ 1], A1[ka ^ 1]);
        if (idx + 1 < 64) LOAD_B(idx + 1, B[bb ^ 1]);
        acc0[nt] = __builtin_amdgcn_wmma_f32_16x16x32_bf16(
            false, A0[ka].v, false, B[bb].v, (short)0, acc0[nt], false, false);
        acc1[nt] = __builtin_amdgcn_wmma_f32_16x16x32_bf16(
            false, A1[ka].v, false, B[bb].v, (short)0, acc1[nt], false, false);
    }
#undef LOAD_A
#undef LOAD_B

    // Epilogue: hid = relu(acc + b1), s = hid . W2 partial per lane.
    // C/D layout: tile nt, VGPR r, lanes 0-15 -> (row r, col nt*16+lane),
    // lanes 16-31 -> (row r+8, col nt*16+lane-16).
    float s0[8], s1[8];
#pragma unroll
    for (int r = 0; r < 8; ++r) { s0[r] = 0.f; s1[r] = 0.f; }
#pragma unroll
    for (int nt = 0; nt < 8; ++nt) {
#pragma unroll
        for (int r = 0; r < 8; ++r) {
            s0[r] += fmaxf(acc0[nt][r] + b1v[nt], 0.f) * w2v[nt];
            s1[r] += fmaxf(acc1[nt][r] + b1v[nt], 0.f) * w2v[nt];
        }
    }
    // Reduce across the 16 lanes of each half (xor of bits 0..3 stays in-half).
#pragma unroll
    for (int mask = 1; mask <= 8; mask <<= 1) {
#pragma unroll
        for (int r = 0; r < 8; ++r) {
            s0[r] += __shfl_xor(s0[r], mask, 32);
            s1[r] += __shfl_xor(s1[r], mask, 32);
        }
    }

    const float bias2 = b2[0];
    if (m < 8) {
        float v0 = s0[0], v1 = s1[0];
#pragma unroll
        for (int j = 1; j < 8; ++j) {
            v0 = (m == j) ? s0[j] : v0;
            v1 = (m == j) ? s1[j] : v1;
        }
        // half 0 holds rows 0..7 (r = register index), half 1 rows 8..15
        long long row0 = base + half * 8 + m;        // M-tile 0
        long long row1 = base + 16 + half * 8 + m;   // M-tile 1
        if (row0 < E) out[row0] = 1.f / (1.f + __expf(-(v0 + bias2)));
        if (row1 < E) out[row1] = 1.f / (1.f + __expf(-(v1 + bias2)));
    }
}

extern "C" void kernel_launch(void* const* d_in, const int* in_sizes, int n_in,
                              void* d_out, int out_size, void* d_ws, size_t ws_size,
                              hipStream_t stream) {
    (void)n_in; (void)out_size; (void)ws_size;
    const float* h   = (const float*)d_in[0];
    const int*   ei  = (const int*)d_in[1];
    const float* W1  = (const float*)d_in[2];
    const float* b1  = (const float*)d_in[3];
    const float* W2  = (const float*)d_in[4];
    const float* b2  = (const float*)d_in[5];
    float* out = (float*)d_out;

    const int nh = in_sizes[0];           // N_NODES * 128
    const int E  = in_sizes[1] / 2;       // edges

    __bf16* h_bf = (__bf16*)d_ws;
    size_t  off  = ((size_t)nh * sizeof(__bf16) + 255) & ~(size_t)255;
    __bf16* w1p  = (__bf16*)((char*)d_ws + off);

    convert_h_bf16<<<(nh + 255) / 256, 256, 0, stream>>>(h, h_bf, nh);
    pack_w1<<<(64 * 32 * 16) / 256, 256, 0, stream>>>(W1, w1p);

    const int blocks = (E + 255) / 256;   // 256 edges per block (8 waves x 32)
    edge_mlp_wmma<<<blocks, 256, 0, stream>>>(h_bf, ei, w1p, b1, W2, b2, out, E);
}
